// SelfAttention_20830591386009
// MI455X (gfx1250) — compile-verified, matrix operand order checked by reading, not run on hardware
//
#include <hip/hip_runtime.h>
#include <hip/hip_bf16.h>
#include <hip/hip_fp16.h>

typedef _Float16 h16;
typedef __attribute__((ext_vector_type(16))) _Float16 v16h;
typedef __attribute__((ext_vector_type(8)))  float    v8f;

#define DEV __device__ __forceinline__

// Problem constants (match reference)
constexpr int BQ    = 4;      // batch
constexpr int NSEQ  = 2048;   // sequence length
constexpr int DHEAD = 256;    // per-head dim (== model dim D)
constexpr int NH    = 8;      // heads
constexpr int DMODEL= 256;    // input feature dim
constexpr int DALL  = DHEAD * NH;   // 2048
constexpr int MTOT  = BQ * NSEQ;    // 8192

// ---------------------------------------------------------------------------
// CDNA5 async global->LDS copy (ASYNCcnt-tracked DMA, bypasses VGPRs).
// GV addressing: per-lane 64-bit global address; VDST holds the LDS byte
// address.  Low 32 bits of a generic __shared__ pointer are the LDS offset
// (flat-address aperture truncation, ISA 10.2), so no addrspacecast needed.
// ---------------------------------------------------------------------------
DEV void async_b128(const void* gsrc, void* ldst) {
  unsigned long long ga = (unsigned long long)(uintptr_t)gsrc;
  unsigned int       la = (unsigned int)(uintptr_t)ldst;
  asm volatile("global_load_async_to_lds_b128 %0, %1, off"
               :: "v"(la), "v"(ga) : "memory");
}
DEV void wait_async0() {
  asm volatile("s_wait_asynccnt 0x0" ::: "memory");
}

// ---------------------------------------------------------------------------
// WMMA helpers.  v_wmma_f32_16x16x32_f16: A 16x32 (MxK), B 32x16 (KxN),
// C/D 16x16 f32.  A-operand layout (ISA 7.12.2):
//   lanes 0-15  : row = lane,    K = {0..7, 16..23}
//   lanes 16-31 : row = lane-16, K = {8..15, 24..31}
// Both operand tiles are kept row-major-along-K in LDS (B tiles transposed
// as [n][k]), so a fragment load is two contiguous b128 LDS reads.
// C/D layout: vgpr r holds (M = r + 8*(lane>=16), N = lane%16).
// ---------------------------------------------------------------------------
DEV v16h load_frag(const h16* base, int ld) {
  const int lane = threadIdx.x & 31;
  const int row  = lane & 15;
  const int sel  = (lane >> 4) << 3;   // 0 or 8
  const h16* p = base + (size_t)row * ld + sel;
  union { v16h v; uint4 u[2]; } f;
  f.u[0] = *(const uint4*)(p);         // K = sel .. sel+7
  f.u[1] = *(const uint4*)(p + 16);    // K = 16+sel .. 16+sel+7
  return f.v;
}

DEV v8f wmma16(v16h a, v16h b, v8f c) {
  return __builtin_amdgcn_wmma_f32_16x16x32_f16(
      false, a, false, b, (short)0, c, false, false);
}

// ---------------------------------------------------------------------------
// fp32 -> fp16 convert (with folded scale), vectorized x4
// ---------------------------------------------------------------------------
__global__ void k_cvt(const float* __restrict__ src, h16* __restrict__ dst,
                      int n4, float scale) {
  int i = blockIdx.x * blockDim.x + threadIdx.x;
  if (i >= n4) return;
  float4 f = ((const float4*)src)[i];
  union { uint2 u; h16 h[4]; } o;
  o.h[0] = (h16)(f.x * scale);
  o.h[1] = (h16)(f.y * scale);
  o.h[2] = (h16)(f.z * scale);
  o.h[3] = (h16)(f.w * scale);
  ((uint2*)dst)[i] = o.u;
}

// fp32 [K][N] -> fp16 transposed [N][K]  (one-time weight prep so GEMM
// B-tiles become straight strided copies, i.e. async-DMA-able)
__global__ void k_cvt_t(const float* __restrict__ src, h16* __restrict__ dst,
                        int K, int N, float scale) {
  int i = blockIdx.x * blockDim.x + threadIdx.x;
  if (i >= K * N) return;
  int k = i / N;
  int n = i - k * N;
  dst[(size_t)n * K + k] = (h16)(src[i] * scale);
}

// ---------------------------------------------------------------------------
// Tiled WMMA GEMM:  C[M,Nc] = A[M,K] * B[K,Nc], B supplied TRANSPOSED [Nc][K].
// Block tile 128x128, K-step 32, 256 threads = 8 waves, wave tile 64x32.
// Double-buffered LDS; tiles staged with async global->LDS DMA and overlapped
// with WMMA compute (issue next tile -> compute -> s_wait_asynccnt -> barrier).
// MODE 0: fp16 out row-major; MODE 1: fp16 [b,h,n,d]; MODE 3: fp16 [b,h,d,n];
// MODE 2: f32 out + bias.
// ---------------------------------------------------------------------------
template <int MODE>
__global__ __launch_bounds__(256) void k_gemm(const h16* __restrict__ A,
                                              const h16* __restrict__ BmT,
                                              void* __restrict__ Cout,
                                              const float* __restrict__ bias,
                                              int M, int Nc, int K) {
  __shared__ h16 As[2][128 * 32];   // [m][k]
  __shared__ h16 Bs[2][128 * 32];   // [n][k]
  const int tid  = threadIdx.x;
  const int m0   = blockIdx.y * 128;
  const int n0   = blockIdx.x * 128;
  const int wave = tid >> 5;
  const int wm   = wave >> 2;    // 0..1
  const int wn   = wave & 3;     // 0..3

  auto fill = [&](int kk, int buf) {
#pragma unroll
    for (int it = 0; it < 2; ++it) {
      int c = tid + it * 256;          // 0..511 chunks of 8 fp16
      int r = c >> 2;                  // tile row 0..127
      int s = (c & 3) << 3;            // k-offset {0,8,16,24}
      async_b128(&A  [(size_t)(m0 + r) * K + kk + s], &As[buf][r * 32 + s]);
      async_b128(&BmT[(size_t)(n0 + r) * K + kk + s], &Bs[buf][r * 32 + s]);
    }
  };

  const v8f vzero = {0.f, 0.f, 0.f, 0.f, 0.f, 0.f, 0.f, 0.f};
  v8f acc[4][2];
#pragma unroll
  for (int i = 0; i < 4; ++i)
#pragma unroll
    for (int j = 0; j < 2; ++j) acc[i][j] = vzero;

  fill(0, 0);
  wait_async0();
  __syncthreads();

  for (int kk = 0, t = 0; kk < K; kk += 32, ++t) {
    const int cur = t & 1;
    if (kk + 32 < K) fill(kk + 32, cur ^ 1);   // DMA next tile during compute

    v16h af[4], bf[2];
#pragma unroll
    for (int i = 0; i < 4; ++i)
      af[i] = load_frag(&As[cur][(wm * 64 + i * 16) * 32], 32);
#pragma unroll
    for (int j = 0; j < 2; ++j)
      bf[j] = load_frag(&Bs[cur][(wn * 32 + j * 16) * 32], 32);
#pragma unroll
    for (int i = 0; i < 4; ++i)
#pragma unroll
      for (int j = 0; j < 2; ++j) acc[i][j] = wmma16(af[i], bf[j], acc[i][j]);

    wait_async0();
    __syncthreads();
  }

  // ---- epilogue ----
  const int lane = tid & 31;
  const int rh   = (lane >> 4) << 3;   // 0 or 8
  const int ncl  = lane & 15;
#pragma unroll
  for (int i = 0; i < 4; ++i)
#pragma unroll
    for (int j = 0; j < 2; ++j)
#pragma unroll
      for (int r = 0; r < 8; ++r) {
        int m = m0 + wm * 64 + i * 16 + r + rh;
        int n = n0 + wn * 32 + j * 16 + ncl;
        float v = acc[i][j][r];
        if (MODE == 0) {
          ((h16*)Cout)[(size_t)m * Nc + n] = (h16)v;
        } else if (MODE == 1) {
          int bb = m >> 11, nr = m & (NSEQ - 1);
          int hh = n >> 8,  dd = n & (DHEAD - 1);
          ((h16*)Cout)[(((size_t)(bb * NH + hh)) * NSEQ + nr) * DHEAD + dd] = (h16)v;
        } else if (MODE == 3) {
          int bb = m >> 11, nr = m & (NSEQ - 1);
          int hh = n >> 8,  dd = n & (DHEAD - 1);
          ((h16*)Cout)[(((size_t)(bb * NH + hh)) * DHEAD + dd) * NSEQ + nr] = (h16)v;
        } else {
          ((float*)Cout)[(size_t)m * Nc + n] = v + bias[n];
        }
      }
}

// ---------------------------------------------------------------------------
// Fused flash attention per (b, h).
//   grid = (NSEQ/128, NH, BQ), 256 threads = 8 waves; each wave owns 16 query
//   rows with Q fragments (16x256 fp16 = 64 VGPRs) resident.
//   Double-buffered 64-row K/V tiles staged via async DMA; V arrives
//   pre-transposed [d][n] == B-operand [n][k] layout.  Dedicated P-staging
//   buffer -> single barrier per K-block.
//   Dynamic LDS layout (all offsets computed arithmetically; no pointer
//   arrays into LDS -- those lower to unsupported static initializers):
//     [    0) Ks buf0  64x256
//     [16384) Ks buf1
//     [32768) Vs buf0  256x64
//     [49152) Vs buf1
//     [65536) P        8 x 16x64
//   Total 144KB (< 320KB per WGP).
// ---------------------------------------------------------------------------
__global__ __launch_bounds__(256) void k_attn(const h16* __restrict__ Q,
                                              const h16* __restrict__ Kg,
                                              const h16* __restrict__ Vt,
                                              h16* __restrict__ Ctx) {
  extern __shared__ h16 smem[];

  const int tid  = threadIdx.x;
  const int wave = tid >> 5;
  const int lane = tid & 31;
  const int rh   = (lane >> 4) << 3;
  const int ncl  = lane & 15;
  const int bb   = blockIdx.z;
  const int hh   = blockIdx.y;
  const int q0   = blockIdx.x * 128 + wave * 16;

  const h16* Qbase = Q  + ((size_t)(bb * NH + hh) * NSEQ + q0) * DHEAD;
  const h16* Kbase = Kg + (size_t)(bb * NH + hh) * NSEQ * DHEAD;
  const h16* Vbase = Vt + (size_t)(bb * NH + hh) * DHEAD * NSEQ;

  auto fill = [&](int kb, int buf) {
    h16* Ksf = smem + buf * 16384;           // [64][256]
    h16* Vsf = smem + 32768 + buf * 16384;   // [256][64]
    const int k0 = kb * 64;
#pragma unroll
    for (int it = 0; it < 8; ++it) {
      int c  = tid + it * 256;         // 0..2047 chunks of 8 fp16
      int kr = c >> 5;                 // K row 0..63
      int ks = (c & 31) << 3;
      async_b128(&Kbase[(size_t)(k0 + kr) * DHEAD + ks], &Ksf[kr * 256 + ks]);
      int dr = c >> 3;                 // V^T row (d) 0..255
      int ds = (c & 7) << 3;
      async_b128(&Vbase[(size_t)dr * NSEQ + k0 + ds], &Vsf[dr * 64 + ds]);
    }
  };

  // resident Q fragments: 8 x (16x32)
  v16h qf[8];
#pragma unroll
  for (int f = 0; f < 8; ++f) qf[f] = load_frag(Qbase + f * 32, DHEAD);

  const v8f vzero = {0.f, 0.f, 0.f, 0.f, 0.f, 0.f, 0.f, 0.f};
  v8f acc[16];                    // context accumulator 16x256 f32
#pragma unroll
  for (int t = 0; t < 16; ++t) acc[t] = vzero;
  float mstate[8], lstate[8];
#pragma unroll
  for (int r = 0; r < 8; ++r) { mstate[r] = -1e30f; lstate[r] = 0.f; }

  fill(0, 0);
  wait_async0();
  __syncthreads();

  constexpr int NB = NSEQ / 64;
  for (int kb = 0; kb < NB; ++kb) {
    const int cur = kb & 1;
    const h16* Ks = smem + cur * 16384;
    const h16* Vs = smem + 32768 + cur * 16384;
    if (kb + 1 < NB) fill(kb + 1, cur ^ 1);   // DMA next K/V block during compute

    // ---- S = Q K^T : 4 tiles of 16x16, accumulate over d = 256 ----
    v8f s[4];
#pragma unroll
    for (int j = 0; j < 4; ++j) s[j] = vzero;
#pragma unroll
    for (int j = 0; j < 4; ++j)
#pragma unroll
      for (int kc = 0; kc < 8; ++kc) {
        v16h bf = load_frag(&Ks[(j * 16) * 256 + kc * 32], 256);
        s[j] = wmma16(qf[kc], bf, s[j]);
      }

    // ---- online softmax (row stats via width-16 shuffle reductions) ----
    float sc[8];
#pragma unroll
    for (int r = 0; r < 8; ++r) {
      float mv = fmaxf(fmaxf(s[0][r], s[1][r]), fmaxf(s[2][r], s[3][r]));
      mv = fmaxf(mv, __shfl_xor(mv, 1, 32));
      mv = fmaxf(mv, __shfl_xor(mv, 2, 32));
      mv = fmaxf(mv, __shfl_xor(mv, 4, 32));
      mv = fmaxf(mv, __shfl_xor(mv, 8, 32));
      float mnew = fmaxf(mstate[r], mv);
      sc[r] = __expf(mstate[r] - mnew);
      mstate[r] = mnew;
      float rs = 0.f;
#pragma unroll
      for (int j = 0; j < 4; ++j) {
        float e = __expf(s[j][r] - mnew);
        s[j][r] = e;
        rs += e;
      }
      rs += __shfl_xor(rs, 1, 32);
      rs += __shfl_xor(rs, 2, 32);
      rs += __shfl_xor(rs, 4, 32);
      rs += __shfl_xor(rs, 8, 32);
      lstate[r] = lstate[r] * sc[r] + rs;
    }
#pragma unroll
    for (int t = 0; t < 16; ++t)
#pragma unroll
      for (int r = 0; r < 8; ++r) acc[t][r] *= sc[r];

    // ---- stage P (16x64 fp16) in the per-wave P buffer ----
    h16* Pw = smem + 65536 + wave * (16 * 64);
#pragma unroll
    for (int j = 0; j < 4; ++j)
#pragma unroll
      for (int r = 0; r < 8; ++r)
        Pw[(r + rh) * 64 + j * 16 + ncl] = (h16)s[j][r];
    // wave-local RAW on LDS: DS ops from one wave complete in order.

    // ---- ctx += P V : 16 d-tiles, k = 64 ----
    v16h p0 = load_frag(Pw, 64);
    v16h p1 = load_frag(Pw + 32, 64);
#pragma unroll
    for (int dt = 0; dt < 16; ++dt) {
      v16h b0 = load_frag(&Vs[dt * 16 * 64], 64);
      v16h b1 = load_frag(&Vs[dt * 16 * 64 + 32], 64);
      acc[dt] = wmma16(p0, b0, acc[dt]);
      acc[dt] = wmma16(p1, b1, acc[dt]);
    }

    wait_async0();
    __syncthreads();
  }

  // ---- normalize and emit context [b, n, h*D + d] fp16 ----
  float inv[8];
#pragma unroll
  for (int r = 0; r < 8; ++r) inv[r] = lstate[r] > 0.f ? 1.f / lstate[r] : 0.f;
#pragma unroll
  for (int dt = 0; dt < 16; ++dt)
#pragma unroll
    for (int r = 0; r < 8; ++r) {
      int m = q0 + r + rh;
      size_t idx = ((size_t)(bb * NSEQ + m)) * DALL + hh * DHEAD + dt * 16 + ncl;
      Ctx[idx] = (h16)(acc[dt][r] * inv[r]);
    }
}

// ---------------------------------------------------------------------------
// Launcher.  Workspace map (~136 MiB):
//   [  0MiB) Xh    8192x256 fp16
//   [  4MiB) WqT   [2048][256] fp16 (pre-scaled by 256^-0.25 = 0.25)
//   [  5MiB) WkT   [2048][256] fp16 (pre-scaled)
//   [  6MiB) WvT   [2048][256] fp16
//   [  7MiB) WoT   [256][2048] fp16
//   [  8MiB) Qh    [b,h,n,d]
//   [ 40MiB) Kh    [b,h,n,d]
//   [ 72MiB) Vth   [b,h,d,n]
//   [104MiB) Ctx   [b*n, h*d]
// ---------------------------------------------------------------------------
extern "C" void kernel_launch(void* const* d_in, const int* in_sizes, int n_in,
                              void* d_out, int out_size, void* d_ws, size_t ws_size,
                              hipStream_t stream) {
  const float* X  = (const float*)d_in[0];
  const float* Wq = (const float*)d_in[1];
  const float* Wk = (const float*)d_in[2];
  const float* Wv = (const float*)d_in[3];
  const float* Wo = (const float*)d_in[4];
  const float* bo = (const float*)d_in[5];

  char* ws = (char*)d_ws;
  h16* Xh  = (h16*)(ws);
  h16* WqT = (h16*)(ws + ((size_t)4   << 20));
  h16* WkT = (h16*)(ws + ((size_t)5   << 20));
  h16* WvT = (h16*)(ws + ((size_t)6   << 20));
  h16* WoT = (h16*)(ws + ((size_t)7   << 20));
  h16* Qh  = (h16*)(ws + ((size_t)8   << 20));
  h16* Kh  = (h16*)(ws + ((size_t)40  << 20));
  h16* Vth = (h16*)(ws + ((size_t)72  << 20));
  h16* Ctx = (h16*)(ws + ((size_t)104 << 20));

  const float qk_scale = 0.25f;  // 256^(-1/4)

  int n4 = MTOT * DMODEL / 4;
  k_cvt<<<(n4 + 255) / 256, 256, 0, stream>>>(X, Xh, n4, 1.0f);
  int ne = DMODEL * DALL;
  k_cvt_t<<<(ne + 255) / 256, 256, 0, stream>>>(Wq, WqT, DMODEL, DALL, qk_scale);
  k_cvt_t<<<(ne + 255) / 256, 256, 0, stream>>>(Wk, WkT, DMODEL, DALL, qk_scale);
  k_cvt_t<<<(ne + 255) / 256, 256, 0, stream>>>(Wv, WvT, DMODEL, DALL, 1.0f);
  k_cvt_t<<<(ne + 255) / 256, 256, 0, stream>>>(Wo, WoT, DALL, DMODEL, 1.0f);

  dim3 gqkv(DALL / 128, MTOT / 128);   // (16, 64)
  k_gemm<1><<<gqkv, 256, 0, stream>>>(Xh, WqT, Qh,  nullptr, MTOT, DALL, DMODEL);
  k_gemm<1><<<gqkv, 256, 0, stream>>>(Xh, WkT, Kh,  nullptr, MTOT, DALL, DMODEL);
  k_gemm<3><<<gqkv, 256, 0, stream>>>(Xh, WvT, Vth, nullptr, MTOT, DALL, DMODEL);

  const int attn_lds = (2 * (16384 + 16384) + 8192) * (int)sizeof(h16); // 144KB
  k_attn<<<dim3(NSEQ / 128, NH, BQ), 256, attn_lds, stream>>>(Qh, Kh, Vth, Ctx);

  dim3 gout(DMODEL / 128, MTOT / 128); // (2, 64)
  k_gemm<2><<<gout, 256, 0, stream>>>(Ctx, WoT, d_out, bo, MTOT, DMODEL, DALL);

  (void)in_sizes; (void)n_in; (void)out_size; (void)ws_size;
}
